// Decoder_14920716386787
// MI455X (gfx1250) — compile-verified, matrix operand order checked by reading, not run on hardware
//
#include <hip/hip_runtime.h>
#include <hip/hip_bf16.h>
#include <stdint.h>

typedef unsigned short u16;
typedef __bf16 bf16x16 __attribute__((ext_vector_type(16)));
typedef float  v8f     __attribute__((ext_vector_type(8)));
typedef u16    v16u16  __attribute__((ext_vector_type(16)));
typedef unsigned int u32x4 __attribute__((ext_vector_type(4)));
typedef int          i32x8 __attribute__((ext_vector_type(8)));
typedef int          i32x4 __attribute__((ext_vector_type(4)));

#define H_   256
#define LAT_ 128
#define D_   512
#define B_   256
#define G4H  1024
#define NWG  16
#define LDS_BYTES (160 * 1024)

// ---------- small device helpers ----------

__device__ __forceinline__ u16 f2bf(float f) {
  unsigned u = __float_as_uint(f);
  unsigned r = u + 0x7FFFu + ((u >> 16) & 1u);   // round-to-nearest-even
  return (u16)(r >> 16);
}
// v_exp_f32 + v_rcp_f32 instead of IEEE division chains
__device__ __forceinline__ float sigf(float x) {
  return __builtin_amdgcn_rcpf(1.0f + __expf(-x));
}
__device__ __forceinline__ float tanh_(float x) {
  return 1.0f - 2.0f * __builtin_amdgcn_rcpf(__expf(2.0f * x) + 1.0f);
}

__device__ __forceinline__ v8f wmma_bf16(bf16x16 a, bf16x16 b, v8f c) {
  // (neg_a, A, neg_b, B, c_mod, C, reuse_a, reuse_b)
  return __builtin_amdgcn_wmma_f32_16x16x32_bf16(false, a, false, b, (short)0, c, false, false);
}

// 16(row) x 32(K) bf16 fragment; base points at (row0, k=0), ld in u16 units.
__device__ __forceinline__ bf16x16 load_frag(const u16* base, int ld, int k0) {
  const int lane = threadIdx.x & 31;
  const u16* p = base + (size_t)(lane & 15) * ld + (k0 + ((lane >> 4) << 4));
  return __builtin_bit_cast(bf16x16, *(const v16u16*)p);
}

// 16x16 f32 accumulator-init tile from row-major [.., ld] (C/D WMMA layout).
__device__ __forceinline__ v8f load_ctile(const float* base, int m0, int n0, int ld) {
  const int lane = threadIdx.x & 31;
  const float* p = base + (size_t)(m0 + ((lane >> 4) << 3)) * ld + n0 + (lane & 15);
  v8f c;
#pragma unroll
  for (int r = 0; r < 8; ++r) c[r] = p[(size_t)r * ld];
  return c;
}

// A-fragment of the torch-faithful permuted fc_input:
// row r<128: h0[2r]|c0[2r]|h0[2r+1]|c0[2r+1]; r>=128: same with layer 1.
__device__ __forceinline__ bf16x16 load_frag_fc(const u16* h0b, const u16* c0b,
                                                const u16* h1b, const u16* c1b,
                                                int m0, int k0) {
  const int lane = threadIdx.x & 31;
  int r = m0 + (lane & 15);
  int k = k0 + ((lane >> 4) << 4);
  int layer = r >> 7; r &= 127;
  int seg  = k >> 8;                // 0..3, each 32-wide K-chunk stays in one segment
  int b    = 2 * r + (seg >> 1);
  int offc = k & 255;
  const u16* src = layer ? ((seg & 1) ? c1b : h1b) : ((seg & 1) ? c0b : h0b);
  return __builtin_bit_cast(bf16x16, *(const v16u16*)(src + (size_t)b * H_ + offc));
}

// device-scope sense barrier for the persistent 16-WGP kernel
__device__ __forceinline__ void grid_sync(unsigned* cnt, unsigned* gen, unsigned nblk) {
  __syncthreads();
#if __has_builtin(__builtin_amdgcn_s_cluster_barrier)
  // s_barrier_signal/-wait -3: hardware cluster sync when dispatched as a
  // cluster; architecturally a NOP otherwise (ClusterID==0).
  __builtin_amdgcn_s_cluster_barrier();
#endif
  if (threadIdx.x == 0) {
    __threadfence();
    unsigned g = __hip_atomic_load(gen, __ATOMIC_RELAXED, __HIP_MEMORY_SCOPE_AGENT);
    unsigned a = __hip_atomic_fetch_add(cnt, 1u, __ATOMIC_ACQ_REL, __HIP_MEMORY_SCOPE_AGENT);
    if (a == nblk - 1u) {
      __hip_atomic_store(cnt, 0u, __ATOMIC_RELAXED, __HIP_MEMORY_SCOPE_AGENT);
      __hip_atomic_fetch_add(gen, 1u, __ATOMIC_RELEASE, __HIP_MEMORY_SCOPE_AGENT);
    } else {
      while (__hip_atomic_load(gen, __ATOMIC_ACQUIRE, __HIP_MEMORY_SCOPE_AGENT) == g)
        __builtin_amdgcn_s_sleep(1);
    }
  }
  __syncthreads();
}

#if __has_builtin(__builtin_amdgcn_tensor_load_to_lds)
// TDM: DMA a 2-D tile (rows x rowlen bf16 elements, row stride in elements)
// from global memory into LDS at byte offset lds_off. D# per ISA 08 §8.
__device__ __forceinline__ void tdm_load_2d(unsigned lds_off, const void* gptr,
                                            unsigned rows, unsigned rowlen,
                                            unsigned stride) {
  unsigned long long ga = (unsigned long long)gptr;
  u32x4 g0;
  g0[0] = 1u;                                   // count=1, user descriptor
  g0[1] = lds_off;                              // lds_addr (bytes)
  g0[2] = (unsigned)(ga & 0xffffffffu);         // global_addr[31:0]
  g0[3] = (unsigned)((ga >> 32) & 0x01ffffffu)  // global_addr[56:32]
        | (2u << 30);                           // type=2 ("image")
  i32x8 g1;
  g1[0] = (int)(1u << 16);                      // data_size=1 -> 2 bytes/elt
  g1[1] = (int)((rowlen & 0xffffu) << 16);      // tensor_dim0[15:0] @ bits 63:48
  g1[2] = (int)((rows & 0xffffu) << 16);        // tensor_dim1[15:0] @ bits 95:80
  g1[3] = (int)((rowlen & 0xffffu) << 16);      // tile_dim0 @ bits 127:112
  g1[4] = (int)(rows & 0xffffu);                // tile_dim1 @ bits 143:128
  g1[5] = (int)stride;                          // tensor_dim0_stride[31:0]
  g1[6] = 0;
  g1[7] = 0;
  i32x4 z4 = {0, 0, 0, 0};
#if __clang_major__ >= 23
  i32x8 z8 = {0, 0, 0, 0, 0, 0, 0, 0};
  __builtin_amdgcn_tensor_load_to_lds(g0, g1, z4, z4, z8, 0);
#else
  __builtin_amdgcn_tensor_load_to_lds(g0, g1, z4, z4, 0);
#endif
}
#endif

// output projection for timestep tp (each WGP owns 32 of the 512 output columns)
__device__ __forceinline__ void out_proj(int tp, const u16* H0, const u16* C0,
                                         const u16* H1, const u16* C1,
                                         const u16* sWm, const float* bmap,
                                         float* out, int wg, int wv) {
  const int lane = threadIdx.x & 31;
  float* op = out + (size_t)tp * B_ * D_;
  for (int mi = 0; mi < 2; ++mi) {
    const int m0 = (wv * 2 + mi) * 16;
    for (int nt = 0; nt < 2; ++nt) {
      const int nc = wg * 32 + nt * 16;
      float bv = bmap[nc + (lane & 15)];
      v8f acc;
#pragma unroll
      for (int r = 0; r < 8; ++r) acc[r] = bv;
#pragma unroll 4
      for (int k0 = 0; k0 < G4H; k0 += 32) {
        bf16x16 a  = load_frag_fc(H0, C0, H1, C1, m0, k0);
        bf16x16 bw = load_frag(sWm + (size_t)(nt * 16) * G4H, G4H, k0);
        acc = wmma_bf16(a, bw, acc);
      }
#pragma unroll
      for (int r = 0; r < 8; ++r) {
        int m = m0 + r + ((lane >> 4) << 3);
        op[(size_t)m * D_ + nc + (lane & 15)] = sigf(acc[r]);
      }
    }
  }
}

// ---------- prep kernels ----------

__global__ void k_cvt_bf16(const float* __restrict__ src, u16* __restrict__ dst, int n) {
  for (int i = blockIdx.x * blockDim.x + threadIdx.x; i < n; i += gridDim.x * blockDim.x)
    dst[i] = f2bf(src[i]);
}

// X0pre = latent @ Wih0^T + bih0 + bhh0   (constant over all 512 steps)
__global__ void k_x0pre(const float* __restrict__ latent, const float* __restrict__ Wih0,
                        const float* __restrict__ bih0, const float* __restrict__ bhh0,
                        const float* __restrict__ bih1, const float* __restrict__ bhh1,
                        float* __restrict__ X0pre, float* __restrict__ bias1) {
  int idx = blockIdx.x * blockDim.x + threadIdx.x;   // 256*1024 threads
  int b = idx >> 10, n = idx & 1023;
  float s = bih0[n] + bhh0[n];
  const float* lp = latent + (size_t)b * LAT_;
  const float* wp = Wih0 + (size_t)n * LAT_;
#pragma unroll 4
  for (int k = 0; k < LAT_; ++k) s = fmaf(lp[k], wp[k], s);
  X0pre[idx] = s;
  if (b == 0) bias1[n] = bih1[n] + bhh1[n];
}

__global__ void k_init_state(u16* __restrict__ h0buf, u16* __restrict__ h1buf,
                             float* __restrict__ c0f, float* __restrict__ c1f,
                             unsigned* __restrict__ bar) {
  int i = blockIdx.x * blockDim.x + threadIdx.x;     // 65536 threads
  if (i < B_ * H_) {
    h0buf[B_ * H_ + i] = 0;   // parity-1 buffers hold the t = -1 (zero) state
    h1buf[B_ * H_ + i] = 0;
    c0f[i] = 0.0f;
    c1f[i] = 0.0f;
  }
  if (i < 2) bar[i] = 0u;
}

// ---------- persistent LSTM decoder ----------

__global__ void __launch_bounds__(256)
k_decoder_lstm(const u16* __restrict__ Whh0b, const u16* __restrict__ Wih1b,
               const u16* __restrict__ Whh1b, const u16* __restrict__ Wmapb,
               const float* __restrict__ X0pre, const float* __restrict__ bias1,
               const float* __restrict__ bmap, float* __restrict__ out,
               u16* __restrict__ h0buf, u16* __restrict__ h1buf,
               u16* __restrict__ c0buf, u16* __restrict__ c1buf,
               float* __restrict__ c0f, float* __restrict__ c1f,
               unsigned* __restrict__ bar, const int* __restrict__ seqlen) {
  extern __shared__ u16 lds[];
  const int wg   = blockIdx.x;       // 16 WGPs
  const int tid  = threadIdx.x;      // 256 threads = 8 wave32
  const int lane = tid & 31;
  const int wv   = tid >> 5;
  const int j0   = wg * 16;          // owned h-column block per layer
  const int T    = seqlen[0];

  __builtin_prefetch(Whh0b, 0, 3);   // global_prefetch_b8: warm L2 for weight staging
  __builtin_prefetch(Wmapb, 0, 3);

  // LDS layout (160 KB of the 320 KB/WGP budget), gate-major slices so that
  // B-fragments are contiguous rows.
  u16* sW0  = lds;                   // Whh0 slice: 64 x 256
  u16* sW1i = lds + 1 * 64 * 256;    // Wih1 slice: 64 x 256
  u16* sW1h = lds + 2 * 64 * 256;    // Whh1 slice: 64 x 256
  u16* sWm  = lds + 3 * 64 * 256;    // Wmap slice: 32 x 1024

#if __has_builtin(__builtin_amdgcn_tensor_load_to_lds)
  // Stage the weight slices with the Tensor Data Mover: 4 gate chunks
  // (16 rows x 256) per recurrent weight + one 32 x 1024 Wmap tile.
  if (wv == 0) {
    const unsigned offW0  = 0;
    const unsigned offW1i = 1u * 64 * 256 * 2;
    const unsigned offW1h = 2u * 64 * 256 * 2;
    const unsigned offWm  = 3u * 64 * 256 * 2;
    for (unsigned g = 0; g < 4; ++g) {
      size_t srow = (size_t)(g * H_ + j0) * H_;
      tdm_load_2d(offW0  + g * 16 * H_ * 2, Whh0b + srow, 16, H_, H_);
      tdm_load_2d(offW1i + g * 16 * H_ * 2, Wih1b + srow, 16, H_, H_);
      tdm_load_2d(offW1h + g * 16 * H_ * 2, Whh1b + srow, 16, H_, H_);
    }
    tdm_load_2d(offWm, Wmapb + (size_t)wg * 32 * G4H, 32, G4H, G4H);
    __builtin_amdgcn_s_wait_tensorcnt(0);
  }
  __syncthreads();
#else
  for (int i = tid; i < 64 * 256; i += 256) {
    int row = i >> 8, col = i & 255;
    size_t srcr = (size_t)((row >> 4) * H_ + j0 + (row & 15)) * H_ + col;
    sW0[i]  = Whh0b[srcr];
    sW1i[i] = Wih1b[srcr];
    sW1h[i] = Whh1b[srcr];
  }
  for (int i = tid; i < 32 * 1024; i += 256) {
    int row = i >> 10, col = i & 1023;
    sWm[i] = Wmapb[(size_t)(wg * 32 + row) * G4H + col];
  }
  __syncthreads();
#endif

  unsigned* bcnt = bar;
  unsigned* bgen = bar + 1;

  for (int t = 0; t < T; ++t) {
    const int pc = t & 1, pp = pc ^ 1;
    const u16* h0prev = h0buf + pp * (B_ * H_);
    u16*       h0cur  = h0buf + pc * (B_ * H_);
    const u16* h1prev = h1buf + pp * (B_ * H_);
    u16*       h1cur  = h1buf + pc * (B_ * H_);
    u16*       c0cur  = c0buf + pc * (B_ * H_);
    u16*       c1cur  = c1buf + pc * (B_ * H_);

    // ---- phase 1: output(t-1) (independent) fused with LSTM cell 0 (t) ----
    if (t > 0) {
      const int tp = t - 1, par = tp & 1;
      out_proj(tp, h0buf + par * (B_ * H_), c0buf + par * (B_ * H_),
                   h1buf + par * (B_ * H_), c1buf + par * (B_ * H_),
               sWm, bmap, out, wg, wv);
    }
    for (int mi = 0; mi < 2; ++mi) {
      const int m0 = (wv * 2 + mi) * 16;
      v8f acc[4];
#pragma unroll
      for (int g = 0; g < 4; ++g) acc[g] = load_ctile(X0pre, m0, g * H_ + j0, G4H);
      for (int k0 = 0; k0 < H_; k0 += 32) {
        bf16x16 a = load_frag(h0prev + (size_t)m0 * H_, H_, k0);
#pragma unroll
        for (int g = 0; g < 4; ++g)
          acc[g] = wmma_bf16(a, load_frag(sW0 + (size_t)(g * 16) * H_, H_, k0), acc[g]);
      }
#pragma unroll
      for (int r = 0; r < 8; ++r) {
        int m = m0 + r + ((lane >> 4) << 3);
        int j = j0 + (lane & 15);
        float iv = sigf(acc[0][r]), fv = sigf(acc[1][r]);
        float gv = tanh_(acc[2][r]), ov = sigf(acc[3][r]);
        float cn = fv * c0f[(size_t)m * H_ + j] + iv * gv;
        c0f[(size_t)m * H_ + j]   = cn;
        h0cur[(size_t)m * H_ + j] = f2bf(ov * tanh_(cn));
        c0cur[(size_t)m * H_ + j] = f2bf(cn);
      }
    }
    grid_sync(bcnt, bgen, NWG);

    // ---- phase 2: LSTM cell 1 (needs full h0(t)) ----
    for (int mi = 0; mi < 2; ++mi) {
      const int m0 = (wv * 2 + mi) * 16;
      v8f acc[4];
#pragma unroll
      for (int g = 0; g < 4; ++g) {
        float bv = bias1[g * H_ + j0 + (lane & 15)];
#pragma unroll
        for (int r = 0; r < 8; ++r) acc[g][r] = bv;
      }
      for (int k0 = 0; k0 < H_; k0 += 32) {
        bf16x16 a = load_frag(h0cur + (size_t)m0 * H_, H_, k0);
#pragma unroll
        for (int g = 0; g < 4; ++g)
          acc[g] = wmma_bf16(a, load_frag(sW1i + (size_t)(g * 16) * H_, H_, k0), acc[g]);
      }
      for (int k0 = 0; k0 < H_; k0 += 32) {
        bf16x16 a = load_frag(h1prev + (size_t)m0 * H_, H_, k0);
#pragma unroll
        for (int g = 0; g < 4; ++g)
          acc[g] = wmma_bf16(a, load_frag(sW1h + (size_t)(g * 16) * H_, H_, k0), acc[g]);
      }
#pragma unroll
      for (int r = 0; r < 8; ++r) {
        int m = m0 + r + ((lane >> 4) << 3);
        int j = j0 + (lane & 15);
        float iv = sigf(acc[0][r]), fv = sigf(acc[1][r]);
        float gv = tanh_(acc[2][r]), ov = sigf(acc[3][r]);
        float cn = fv * c1f[(size_t)m * H_ + j] + iv * gv;
        c1f[(size_t)m * H_ + j]   = cn;
        h1cur[(size_t)m * H_ + j] = f2bf(ov * tanh_(cn));
        c1cur[(size_t)m * H_ + j] = f2bf(cn);
      }
    }
    grid_sync(bcnt, bgen, NWG);
  }

  // ---- epilogue: output(T-1) ----
  {
    const int tp = T - 1, par = tp & 1;
    out_proj(tp, h0buf + par * (B_ * H_), c0buf + par * (B_ * H_),
                 h1buf + par * (B_ * H_), c1buf + par * (B_ * H_),
             sWm, bmap, out, wg, wv);
  }
}

// ---------- host ----------

extern "C" void kernel_launch(void* const* d_in, const int* in_sizes, int n_in,
                              void* d_out, int out_size, void* d_ws, size_t ws_size,
                              hipStream_t stream) {
  const float* latent = (const float*)d_in[0];
  const float* Wih0   = (const float*)d_in[1];
  const float* Whh0   = (const float*)d_in[2];
  const float* bih0   = (const float*)d_in[3];
  const float* bhh0   = (const float*)d_in[4];
  const float* Wih1   = (const float*)d_in[5];
  const float* Whh1   = (const float*)d_in[6];
  const float* bih1   = (const float*)d_in[7];
  const float* bhh1   = (const float*)d_in[8];
  const float* Wmap   = (const float*)d_in[9];
  const float* bmap   = (const float*)d_in[10];
  const int*   seqlen = (const int*)d_in[11];

  char* ws = (char*)d_ws;
  size_t off = 0;
  auto carve = [&](size_t bytes) -> void* {
    void* p = ws + off;
    off += (bytes + 255) & ~(size_t)255;
    return p;
  };
  u16*   Whh0b = (u16*)carve((size_t)1024 * 256 * sizeof(u16));
  u16*   Wih1b = (u16*)carve((size_t)1024 * 256 * sizeof(u16));
  u16*   Whh1b = (u16*)carve((size_t)1024 * 256 * sizeof(u16));
  u16*   Wmapb = (u16*)carve((size_t)512 * 1024 * sizeof(u16));
  float* X0pre = (float*)carve((size_t)256 * 1024 * sizeof(float));
  float* bias1 = (float*)carve((size_t)1024 * sizeof(float));
  u16*   h0buf = (u16*)carve((size_t)2 * 256 * 256 * sizeof(u16));
  u16*   h1buf = (u16*)carve((size_t)2 * 256 * 256 * sizeof(u16));
  u16*   c0buf = (u16*)carve((size_t)2 * 256 * 256 * sizeof(u16));
  u16*   c1buf = (u16*)carve((size_t)2 * 256 * 256 * sizeof(u16));
  float* c0f   = (float*)carve((size_t)256 * 256 * sizeof(float));
  float* c1f   = (float*)carve((size_t)256 * 256 * sizeof(float));
  unsigned* bar = (unsigned*)carve(256);

  k_cvt_bf16<<<256, 256, 0, stream>>>(Whh0, Whh0b, 1024 * 256);
  k_cvt_bf16<<<256, 256, 0, stream>>>(Wih1, Wih1b, 1024 * 256);
  k_cvt_bf16<<<256, 256, 0, stream>>>(Whh1, Whh1b, 1024 * 256);
  k_cvt_bf16<<<512, 256, 0, stream>>>(Wmap, Wmapb, 512 * 1024);
  k_x0pre<<<1024, 256, 0, stream>>>(latent, Wih0, bih0, bhh0, bih1, bhh1, X0pre, bias1);
  k_init_state<<<256, 256, 0, stream>>>(h0buf, h1buf, c0f, c1f, bar);

  k_decoder_lstm<<<NWG, 256, LDS_BYTES, stream>>>(
      Whh0b, Wih1b, Whh1b, Wmapb, X0pre, bias1, bmap, (float*)d_out,
      h0buf, h1buf, c0buf, c1buf, c0f, c1f, bar, seqlen);
}
